// QIFNetwork_24000277250381
// MI455X (gfx1250) — matrix operand order repaired and checked by reading, not compiled, not run on hardware
//
#include <hip/hip_runtime.h>
#include <hip/hip_bf16.h>
#include <cstdint>

// ---------------- constants (match reference) ----------------
#define TSTEPS  50
#define DT_STEP 0.05f
#define VTH     1.0f

// ---------------- types for WMMA ----------------
typedef __attribute__((ext_vector_type(16))) __bf16        v16bf;
typedef __attribute__((ext_vector_type(8)))  float         v8f;
typedef __attribute__((ext_vector_type(4)))  unsigned int  u32x4;

union Frag {
  v16bf v;
  u32x4 q[2];
};

// ---------------- helpers ----------------
__device__ __forceinline__ unsigned short f32_to_bf16_rne(float f) {
  unsigned int u = __float_as_uint(f);
  u += 0x7FFFu + ((u >> 16) & 1u);           // round-to-nearest-even
  return (unsigned short)(u >> 16);
}

__device__ __forceinline__ float hash_u01(unsigned int x) {
  x ^= x >> 17; x *= 0xED5AD4BBu;
  x ^= x >> 11; x *= 0xAC4C1B51u;
  x ^= x >> 15; x *= 0x31848BABu;
  x ^= x >> 14;
  return (float)(x >> 8) * (1.0f / 16777216.0f);
}

__device__ __forceinline__ void mma_all(v8f acc[2][2], const Frag a[2], const Frag b[2]) {
#pragma unroll
  for (int m = 0; m < 2; ++m)
#pragma unroll
    for (int t = 0; t < 2; ++t)
      acc[m][t] = __builtin_amdgcn_wmma_f32_16x16x32_bf16(
          false, a[m].v, false, b[t].v, (short)0, acc[m][t], false, false);
}

// ---------------- WMMA GEMM core: one wave computes 32(M) x 32(N) ----------------
// Sized to fit the 128-VGPR allocation the compiler gives these kernels:
//   acc 32 + A frags 32 + B frags 32 + addresses ~8  -> no scratch spills.
// K is a compile-time constant so the 2nd A row and 2nd B column are immediate
// offsets (+16*K elems, < 2^23 bytes) off just TWO base addresses.
// Double-buffered fragments hide load latency under the WMMAs.
template <int K>
__device__ __forceinline__ void gemm_core_32x32(
    const unsigned short* __restrict__ A,
    const unsigned short* __restrict__ Bt,
    int m0, int n0, v8f acc[2][2])
{
  const int lane = threadIdx.x & 31;
  const int hi   = lane >> 4;
  const int lo   = lane & 15;

  // A: bf16 [M,K]; 16x32 A layout -> per lane two 16B chunks at K offsets
  // {8*hi, 16+8*hi}. Row m0+lo (m-tile 0); m-tile 1 at constant +16*K.
  const unsigned short* abase = A + (size_t)(m0 + lo) * K + 8 * hi;
  // B: bf16 [N,K] (pre-transposed); 32x16 B layout -> per lane two 16B chunks at K offsets
  // {8*hi, 16+8*hi}. Column n0+lo; column group t at +t*16*K.
  const unsigned short* bbase = Bt + (size_t)(n0 + lo) * K + 8 * hi;

  Frag a0[2], a1[2], b0[2], b1[2];

  auto load_a2 = [&](Frag a[2], int kk) {
#pragma unroll
    for (int m = 0; m < 2; ++m) {
      a[m].q[0] = *reinterpret_cast<const u32x4*>(abase + m * 16 * K + kk);
      a[m].q[1] = *reinterpret_cast<const u32x4*>(abase + m * 16 * K + kk + 16);
    }
  };
  auto load_b2 = [&](Frag b[2], int kk) {
#pragma unroll
    for (int t = 0; t < 2; ++t) {
      b[t].q[0] = *reinterpret_cast<const u32x4*>(bbase + t * 16 * K + kk);
      b[t].q[1] = *reinterpret_cast<const u32x4*>(bbase + t * 16 * K + kk + 16);
    }
  };

  // prologue: stage K-step 0 into buffer 0
  load_a2(a0, 0);
  load_b2(b0, 0);

  constexpr int S = K >> 5;   // number of 32-wide K steps (even)
  int kk = 32;
#pragma unroll 1
  for (int s = 0; s < S - 2; s += 2) {
    load_a2(a1, kk);
    load_b2(b1, kk);
    mma_all(acc, a0, b0);
    kk += 32;
    load_a2(a0, kk);
    load_b2(b0, kk);
    mma_all(acc, a1, b1);
    kk += 32;
  }
  // epilogue: final staged step + last step
  load_a2(a1, kk);
  load_b2(b1, kk);
  mma_all(acc, a0, b0);
  mma_all(acc, a1, b1);
}

// ---------------- elementwise kernels ----------------
__global__ void k_zero(float* __restrict__ p, int n) {
  int i = blockIdx.x * blockDim.x + threadIdx.x;
  if (i < n) p[i] = 0.0f;
}

__global__ void k_cast_bf16(const float* __restrict__ s, unsigned short* __restrict__ d, int n) {
  int i = blockIdx.x * blockDim.x + threadIdx.x;
  if (i < n) d[i] = f32_to_bf16_rne(s[i]);
}

// src: f32 [K, N] row-major -> dst: bf16 [N, K] row-major (one-time cost per launch)
__global__ void k_transpose_bf16(const float* __restrict__ s, unsigned short* __restrict__ d,
                                 int K, int N) {
  int i = blockIdx.x * blockDim.x + threadIdx.x;
  if (i < K * N) {
    int k = i / N, n = i - k * N;
    d[(size_t)n * K + k] = f32_to_bf16_rne(s[i]);
  }
}

__global__ void k_spikegen(const float* __restrict__ P, unsigned short* __restrict__ S,
                           int n, unsigned int seed) {
  int i = blockIdx.x * blockDim.x + threadIdx.x;
  if (i < n) {
    unsigned int h = (unsigned int)i * 0x9E3779B9u ^ (seed * 0x85EBCA6Bu + 0xC2B2AE35u);
    S[i] = (hash_u01(h) < P[i]) ? (unsigned short)0x3F80u : (unsigned short)0u;  // bf16 1.0/0.0
  }
}

__global__ void k_scale(float* __restrict__ p, int n, float f) {
  int i = blockIdx.x * blockDim.x + threadIdx.x;
  if (i < n) p[i] *= f;
}

// ---------------- GEMM + fused epilogues ----------------
// Encoder: P = sigmoid(A @ B + bias)
template <int K>
__global__ void k_gemm_sigmoid(const unsigned short* __restrict__ A,
                               const unsigned short* __restrict__ Bt,
                               const float* __restrict__ bias,
                               float* __restrict__ P, int N) {
  v8f acc[2][2] = {};
  const int m0 = blockIdx.x * 32, n0 = blockIdx.y * 32;
  gemm_core_32x32<K>(A, Bt, m0, n0, acc);
  const int lane = threadIdx.x & 31, hi = lane >> 4, lo = lane & 15;
#pragma unroll
  for (int mt = 0; mt < 2; ++mt)
#pragma unroll
    for (int t = 0; t < 2; ++t) {
      int col = n0 + t * 16 + lo;
      float bb = bias[col];
#pragma unroll
      for (int j = 0; j < 8; ++j) {
        int row = m0 + mt * 16 + j + 8 * hi;     // C layout: VGPR j -> M = j + 8*hi
        float z = acc[mt][t][j] + bb;
        P[(size_t)row * N + col] = 1.0f / (1.0f + __expf(-z));
      }
    }
}

// Hidden QIF layer: v += DT*v^2 + (A@W + b); spike; reset; emit spikes as bf16
template <int K>
__global__ void k_gemm_qif_hidden(const unsigned short* __restrict__ A,
                                  const unsigned short* __restrict__ Bt,
                                  const float* __restrict__ bias,
                                  float* __restrict__ V,
                                  unsigned short* __restrict__ Sout,
                                  int N) {
  v8f acc[2][2] = {};
  const int m0 = blockIdx.x * 32, n0 = blockIdx.y * 32;
  gemm_core_32x32<K>(A, Bt, m0, n0, acc);
  const int lane = threadIdx.x & 31, hi = lane >> 4, lo = lane & 15;
#pragma unroll
  for (int mt = 0; mt < 2; ++mt)
#pragma unroll
    for (int t = 0; t < 2; ++t) {
      int col = n0 + t * 16 + lo;
      float bb = bias[col];
#pragma unroll
      for (int j = 0; j < 8; ++j) {
        int row = m0 + mt * 16 + j + 8 * hi;
        size_t idx = (size_t)row * N + col;
        float v = V[idx];
        v = v + DT_STEP * v * v + acc[mt][t][j] + bb;
        float s = (v >= VTH) ? 1.0f : 0.0f;
        v *= (1.0f - s);
        V[idx] = v;
        Sout[idx] = (s != 0.0f) ? (unsigned short)0x3F80u : (unsigned short)0u;
      }
    }
}

// Output QIF layer: same dynamics, accumulate spikes into Cnt (d_out)
template <int K>
__global__ void k_gemm_qif_out(const unsigned short* __restrict__ A,
                               const unsigned short* __restrict__ Bt,
                               const float* __restrict__ bias,
                               float* __restrict__ V,
                               float* __restrict__ Cnt,
                               int N) {
  v8f acc[2][2] = {};
  const int m0 = blockIdx.x * 32, n0 = blockIdx.y * 32;
  gemm_core_32x32<K>(A, Bt, m0, n0, acc);
  const int lane = threadIdx.x & 31, hi = lane >> 4, lo = lane & 15;
#pragma unroll
  for (int mt = 0; mt < 2; ++mt)
#pragma unroll
    for (int t = 0; t < 2; ++t) {
      int col = n0 + t * 16 + lo;
      float bb = bias[col];
#pragma unroll
      for (int j = 0; j < 8; ++j) {
        int row = m0 + mt * 16 + j + 8 * hi;
        size_t idx = (size_t)row * N + col;
        float v = V[idx];
        v = v + DT_STEP * v * v + acc[mt][t][j] + bb;
        float s = (v >= VTH) ? 1.0f : 0.0f;
        v *= (1.0f - s);
        V[idx] = v;
        Cnt[idx] += s;     // each (row,col) owned by exactly one lane
      }
    }
}

// ---------------- launch ----------------
extern "C" void kernel_launch(void* const* d_in, const int* in_sizes, int n_in,
                              void* d_out, int out_size, void* d_ws, size_t ws_size,
                              hipStream_t stream) {
  (void)in_sizes; (void)n_in; (void)out_size; (void)ws_size;
  const int B = 512, I = 2048, H = 4096, O = 1024;

  const float* x    = (const float*)d_in[0];
  const float* W_in = (const float*)d_in[1];
  const float* b_in = (const float*)d_in[2];
  const float* W1   = (const float*)d_in[3];
  const float* b1   = (const float*)d_in[4];
  const float* W2   = (const float*)d_in[5];
  const float* b2   = (const float*)d_in[6];
  float* out = (float*)d_out;

  char* wsb = (char*)d_ws;
  size_t off = 0;
  auto wsalloc = [&](size_t bytes) -> char* {
    char* p = wsb + off;
    off += (bytes + 255) & ~(size_t)255;
    return p;
  };
  unsigned short* Wint = (unsigned short*)wsalloc((size_t)I * I * 2);  // W_in^T bf16 [I,I]
  unsigned short* W1t  = (unsigned short*)wsalloc((size_t)I * H * 2);  // W1^T  bf16 [H,I]
  unsigned short* W2t  = (unsigned short*)wsalloc((size_t)H * O * 2);  // W2^T  bf16 [O,H]
  unsigned short* xb   = (unsigned short*)wsalloc((size_t)B * I * 2);  // x bf16 [B,I]
  float*          penc = (float*)wsalloc((size_t)B * I * 4);           // p_enc f32 [B,I]
  unsigned short* spk  = (unsigned short*)wsalloc((size_t)B * I * 2);  // input spikes bf16
  float*          v1   = (float*)wsalloc((size_t)B * H * 4);           // membrane 1
  unsigned short* s1   = (unsigned short*)wsalloc((size_t)B * H * 2);  // layer-1 spikes bf16
  float*          v2   = (float*)wsalloc((size_t)B * O * 4);           // membrane 2

  auto cdiv = [](int a, int b) { return (a + b - 1) / b; };

  // One-time per launch: init state, quantize/transpose weights, cast inputs
  k_zero<<<cdiv(B * H, 256), 256, 0, stream>>>(v1, B * H);
  k_zero<<<cdiv(B * O, 256), 256, 0, stream>>>(v2, B * O);
  k_zero<<<cdiv(B * O, 256), 256, 0, stream>>>(out, B * O);
  k_cast_bf16<<<cdiv(B * I, 256), 256, 0, stream>>>(x, xb, B * I);
  k_transpose_bf16<<<cdiv(I * I, 256), 256, 0, stream>>>(W_in, Wint, I, I);
  k_transpose_bf16<<<cdiv(I * H, 256), 256, 0, stream>>>(W1, W1t, I, H);
  k_transpose_bf16<<<cdiv(H * O, 256), 256, 0, stream>>>(W2, W2t, H, O);

  // Encoder: p_enc = sigmoid(x @ W_in + b_in)
  {
    dim3 g(B / 32, I / 32);
    k_gemm_sigmoid<2048><<<g, 32, 0, stream>>>(xb, Wint, b_in, penc, I);
  }

  // 50 QIF steps
  dim3 g1(B / 32, H / 32);
  dim3 g2(B / 32, O / 32);
  for (int t = 0; t < TSTEPS; ++t) {
    k_spikegen<<<cdiv(B * I, 256), 256, 0, stream>>>(penc, spk, B * I, (unsigned)(t + 1));
    k_gemm_qif_hidden<2048><<<g1, 32, 0, stream>>>(spk, W1t, b1, v1, s1, H);
    k_gemm_qif_out<4096>   <<<g2, 32, 0, stream>>>(s1,  W2t, b2, v2, out, O);
  }

  // rate = count / T
  k_scale<<<cdiv(B * O, 256), 256, 0, stream>>>(out, B * O, 1.0f / (float)TSTEPS);
}
